// RL_agent_47820165874335
// MI455X (gfx1250) — matrix operand order repaired and checked by reading, not compiled, not run on hardware
//
#include <hip/hip_runtime.h>

// ---------------------------------------------------------------------------
// MI455X (gfx1250) implementation: all GEMMs via v_wmma_f32_16x16x32_f16
// (wave32, 16x16 tiles, fp32 accumulate), fp32 VALU for softmax/topk/GRU.
// ---------------------------------------------------------------------------

typedef __attribute__((ext_vector_type(8)))  _Float16 half8;
typedef __attribute__((ext_vector_type(16))) _Float16 v16h;
typedef __attribute__((ext_vector_type(8)))  float    v8f;

namespace cfg {
constexpr int B = 512, S = 128, IN = 64, ACTN = 8, D = 256, H = 4, TOPK = 8, DH = 64;
constexpr int BS = B * S;        // 65536 rows
constexpr int D3 = 3 * D;        // 768
constexpr int KD = TOPK * D;     // 2048
}

static __device__ __forceinline__ v16h load_frag(const _Float16* p0, const _Float16* p1) {
  half8 lo = *(const half8*)p0;
  half8 hi = *(const half8*)p1;
  return __builtin_shufflevector(lo, hi, 0,1,2,3,4,5,6,7,8,9,10,11,12,13,14,15);
}

static __device__ __forceinline__ v8f wmma_f16(v16h a, v16h b, v8f c) {
  // D = A(16x32 f16) * B(32x16 f16) + C(16x16 f32)
  return __builtin_amdgcn_wmma_f32_16x16x32_f16(false, a, false, b, (short)0, c, false, false);
}

static __device__ __forceinline__ v8f v8f_zero() {
  v8f z;
#pragma unroll
  for (int i = 0; i < 8; ++i) z[i] = 0.0f;
  return z;
}

// ---------------------------------------------------------------------------
// Generic f16 GEMM:  C[M,N] = act(A[M,K] * Bt[N,K]^T + bias[N])
// Block tile 128x128, 8 waves, each wave: 16 rows x 128 cols (8 accumulators).
// ---------------------------------------------------------------------------
constexpr int GBM = 128, GBN = 128, GKS = 32, GLS = 40;  // GLS: padded LDS row stride

template <int ACTF, bool S16, bool S32>
__global__ __launch_bounds__(256) void gemm_f16_kernel(
    const _Float16* __restrict__ A, int lda,
    const _Float16* __restrict__ Bt, int ldb,
    const float* __restrict__ bias,
    _Float16* __restrict__ C16, float* __restrict__ C32, int ldc,
    int M, int N, int K) {
  __shared__ _Float16 sA[GBM * GLS];
  __shared__ _Float16 sB[GBN * GLS];
  const int tid  = threadIdx.x;
  const int lane = tid & 31;
  const int wave = tid >> 5;
  const int m0 = blockIdx.y * GBM;
  const int n0 = blockIdx.x * GBN;
  (void)M;

  v8f acc[8];
#pragma unroll
  for (int j = 0; j < 8; ++j) acc[j] = v8f_zero();

  const int mrow = wave * 16 + (lane & 15);
  const int hh   = lane >> 4;

  for (int k0 = 0; k0 < K; k0 += GKS) {
    // Stage A tile (128 rows x 32 halves): 512 x 16B chunks, 2 per thread.
#pragma unroll
    for (int i = 0; i < 2; ++i) {
      int idx = tid + i * 256;
      int row = idx >> 2;
      int kc  = (idx & 3) * 8;
      *(half8*)(sA + row * GLS + kc) =
          *(const half8*)(A + (size_t)(m0 + row) * lda + k0 + kc);
    }
    // Stage Bt tile (128 n-rows x 32 halves), zero-pad beyond N.
#pragma unroll
    for (int i = 0; i < 2; ++i) {
      int idx = tid + i * 256;
      int row = idx >> 2;
      int kc  = (idx & 3) * 8;
      half8 v;
      if (n0 + row < N) {
        v = *(const half8*)(Bt + (size_t)(n0 + row) * ldb + k0 + kc);
      } else {
#pragma unroll
        for (int q = 0; q < 8; ++q) v[q] = (_Float16)0.0f;
      }
      *(half8*)(sB + row * GLS + kc) = v;
    }
    if (k0 + GKS < K)
      __builtin_prefetch(A + (size_t)(m0 + (tid >> 1)) * lda + k0 + GKS, 0, 1);
    __syncthreads();

    // A fragment: lane holds M=lane&15, K halves {hh*16..+15}
    v16h af = load_frag(sA + mrow * GLS + hh * 16,
                        sA + mrow * GLS + hh * 16 + 8);
#pragma unroll
    for (int j = 0; j < 8; ++j) {
      const _Float16* bp = sB + (size_t)(j * 16 + (lane & 15)) * GLS + hh * 16;
      acc[j] = wmma_f16(af, load_frag(bp, bp + 8), acc[j]);
    }
    __syncthreads();
  }

  // C layout: VGPR r -> M = (lane>>4)*8 + r, N = lane&15 within the 16x16 tile.
  const int mb = m0 + wave * 16 + hh * 8;
  const int nl = lane & 15;
#pragma unroll
  for (int j = 0; j < 8; ++j) {
    int gn = n0 + j * 16 + nl;
    if (gn < N) {
      float bv = bias ? bias[gn] : 0.0f;
#pragma unroll
      for (int r = 0; r < 8; ++r) {
        float v = acc[j][r] + bv;
        if (ACTF == 1) v = tanhf(v);
        size_t off = (size_t)(mb + r) * ldc + gn;
        if (S16) C16[off] = (_Float16)v;
        if (S32) C32[off] = v;
      }
    }
  }
}

// ---------------------------------------------------------------------------
// Fused attention per (b,h): scores = QK^T/8, softmax, raw row-sums for topk,
// ctx = attn @ V.  LDS reused (Q/K region -> attn) to stay under 64KB.
// ---------------------------------------------------------------------------
__global__ __launch_bounds__(256) void attn_kernel(
    const _Float16* __restrict__ QKV, _Float16* __restrict__ ctx,
    float* __restrict__ ssum) {
  constexpr int LQ = 72, LV = 136, LA = 136;
  __shared__ _Float16 smem[27136];          // 54,272 bytes
  _Float16* sQ    = smem;                   // [128][72]
  _Float16* sK    = smem + 128 * LQ;        // [128][72]
  _Float16* sVt   = smem + 2 * 128 * LQ;    // [64][136]  (V transposed: [d][k])
  _Float16* sAttn = smem;                   // [128][136] reuses Q/K after barrier

  const int bh = blockIdx.x;
  const int b  = bh >> 2;
  const int h  = bh & 3;
  const int tid  = threadIdx.x;
  const int lane = tid & 31;
  const int wave = tid >> 5;
  const size_t base = (size_t)b * 128 * 768;

#pragma unroll
  for (int i = 0; i < 4; ++i) {
    int idx = tid + i * 256;
    int row = idx >> 3;
    int dc  = (idx & 7) * 8;
    const _Float16* qp = QKV + base + (size_t)row * 768 + h * 64 + dc;
    *(half8*)(sQ + row * LQ + dc) = *(const half8*)qp;
    *(half8*)(sK + row * LQ + dc) = *(const half8*)(qp + 256);
    half8 vv = *(const half8*)(qp + 512);
#pragma unroll
    for (int q = 0; q < 8; ++q) sVt[(size_t)(dc + q) * LV + row] = vv[q];
  }
  __syncthreads();

  const int mrow = wave * 16 + (lane & 15);
  const int hh   = lane >> 4;

  // scores: 128x128, K=64
  v8f acc[8];
#pragma unroll
  for (int j = 0; j < 8; ++j) acc[j] = v8f_zero();
#pragma unroll
  for (int k0 = 0; k0 < 64; k0 += 32) {
    v16h af = load_frag(sQ + mrow * LQ + k0 + hh * 16,
                        sQ + mrow * LQ + k0 + hh * 16 + 8);
#pragma unroll
    for (int j = 0; j < 8; ++j) {
      const _Float16* bp = sK + (size_t)(j * 16 + (lane & 15)) * LQ + k0 + hh * 16;
      acc[j] = wmma_f16(af, load_frag(bp, bp + 8), acc[j]);
    }
  }
  __syncthreads();  // done reading sQ/sK; sAttn may now overwrite them

  const float scale = 0.125f;  // 1/sqrt(64)
#pragma unroll
  for (int j = 0; j < 8; ++j)
#pragma unroll
    for (int r = 0; r < 8; ++r) acc[j][r] *= scale;

  const int srow = wave * 16 + hh * 8;   // + r
  const int ncol = lane & 15;            // + j*16
#pragma unroll
  for (int r = 0; r < 8; ++r) {
    // Row values live in 16 lanes (same hh half) across 8 accumulators.
    float mx = -3.4e38f;
#pragma unroll
    for (int j = 0; j < 8; ++j) mx = fmaxf(mx, acc[j][r]);
#pragma unroll
    for (int o = 1; o < 16; o <<= 1) mx = fmaxf(mx, __shfl_xor(mx, o, 32));
    float sum = 0.0f, raw = 0.0f;
#pragma unroll
    for (int j = 0; j < 8; ++j) {
      raw += acc[j][r];
      float e = __expf(acc[j][r] - mx);
      acc[j][r] = e;
      sum += e;
    }
#pragma unroll
    for (int o = 1; o < 16; o <<= 1) {
      sum += __shfl_xor(sum, o, 32);
      raw += __shfl_xor(raw, o, 32);
    }
    float inv = 1.0f / sum;
#pragma unroll
    for (int j = 0; j < 8; ++j)
      sAttn[(size_t)(srow + r) * LA + j * 16 + ncol] = (_Float16)(acc[j][r] * inv);
    if (ncol == 0)
      ssum[((size_t)h * cfg::B + b) * cfg::S + srow + r] = raw;  // per-head raw sum
  }
  __syncthreads();

  // ctx = attn(128x128) @ V(128x64), Bt = sVt[d][k]
  v8f cacc[4];
#pragma unroll
  for (int j = 0; j < 4; ++j) cacc[j] = v8f_zero();
#pragma unroll
  for (int k0 = 0; k0 < 128; k0 += 32) {
    v16h af = load_frag(sAttn + (size_t)mrow * LA + k0 + hh * 16,
                        sAttn + (size_t)mrow * LA + k0 + hh * 16 + 8);
#pragma unroll
    for (int j = 0; j < 4; ++j) {
      const _Float16* bp = sVt + (size_t)(j * 16 + (lane & 15)) * LV + k0 + hh * 16;
      cacc[j] = wmma_f16(af, load_frag(bp, bp + 8), cacc[j]);
    }
  }
#pragma unroll
  for (int j = 0; j < 4; ++j)
#pragma unroll
    for (int r = 0; r < 8; ++r)
      ctx[((size_t)b * 128 + srow + r) * 256 + h * 64 + j * 16 + ncol] =
          (_Float16)cacc[j][r];
}

// ---------------------------------------------------------------------------
// Top-k (k=8, ties -> lowest index, matching lax.top_k) + gather of trans_out.
// ---------------------------------------------------------------------------
__global__ __launch_bounds__(128) void topk_gather_kernel(
    const float* __restrict__ ssum, const float* __restrict__ TO,
    float* __restrict__ resh_out, _Float16* __restrict__ resh_h) {
  __shared__ float vals[128];
  __shared__ float rv[128];
  __shared__ int   ri[128];
  __shared__ int   sel[8];
  const int b = blockIdx.x;
  const int s = threadIdx.x;
  float v = 0.0f;
#pragma unroll
  for (int h = 0; h < 4; ++h) v += ssum[((size_t)h * 512 + b) * 128 + s];
  vals[s] = v;
  __syncthreads();
  for (int t = 0; t < 8; ++t) {
    rv[s] = vals[s];
    ri[s] = s;
    __syncthreads();
    for (int o = 64; o > 0; o >>= 1) {
      if (s < o) {
        float a = rv[s], c = rv[s + o];
        int ia = ri[s], ic = ri[s + o];
        if (c > a || (c == a && ic < ia)) { rv[s] = c; ri[s] = ic; }
      }
      __syncthreads();
    }
    if (s == 0) { sel[t] = ri[0]; vals[ri[0]] = -3.4e38f; }
    __syncthreads();
  }
  for (int i = s; i < 2048; i += 128) {
    int t = i >> 8, c = i & 255;
    float x = TO[((size_t)b * 128 + sel[t]) * 256 + c];
    resh_out[(size_t)b * 2048 + i] = x;          // fp32 output #2
    resh_h[(size_t)b * 2048 + i]   = (_Float16)x; // f16 copy for GRU GEMM
  }
}

// ---------------------------------------------------------------------------
// GRU gate fusion (pure VALU fp32).
// ---------------------------------------------------------------------------
__global__ __launch_bounds__(256) void gru_kernel(
    const float* __restrict__ gi, const float* __restrict__ gh,
    const float* __restrict__ prev_d, float* __restrict__ out) {
  int i = blockIdx.x * 256 + threadIdx.x;   // B*D elements
  int b = i >> 8;
  int j = i & 255;
  const float* gib = gi + (size_t)b * 768;
  const float* ghb = gh + (size_t)b * 768;
  float r = 1.0f / (1.0f + __expf(-(gib[j] + ghb[j])));
  float z = 1.0f / (1.0f + __expf(-(gib[256 + j] + ghb[256 + j])));
  float n = tanhf(gib[512 + j] + r * ghb[512 + j]);
  out[i] = (1.0f - z) * n + z * prev_d[i];
}

// ---------------------------------------------------------------------------
// Conversion helpers.
// ---------------------------------------------------------------------------
__global__ void cvt_f16_kernel(const float* __restrict__ src,
                               _Float16* __restrict__ dst, int n) {
  int i = blockIdx.x * blockDim.x + threadIdx.x;
  if (i < n) dst[i] = (_Float16)src[i];
}

__global__ void transpose_f16_kernel(const float* __restrict__ src,
                                     _Float16* __restrict__ dst, int K, int N) {
  int i = blockIdx.x * blockDim.x + threadIdx.x;
  if (i < K * N) {
    int k = i / N, n = i - k * N;
    dst[(size_t)n * K + k] = (_Float16)src[i];  // Bt[n][k] = W[k][n]
  }
}

__global__ void fill_aprev_kernel(const float* __restrict__ a_prev,
                                  _Float16* __restrict__ XA) {
  int i = blockIdx.x * blockDim.x + threadIdx.x;
  if (i < cfg::BS * cfg::ACTN) {
    int row = i >> 3, j = i & 7;
    XA[(size_t)row * cfg::D + (cfg::D - cfg::ACTN) + j] = (_Float16)a_prev[i];
  }
}

// ---------------------------------------------------------------------------
// Launcher.
// ---------------------------------------------------------------------------
extern "C" void kernel_launch(void* const* d_in, const int* in_sizes, int n_in,
                              void* d_out, int out_size, void* d_ws, size_t ws_size,
                              hipStream_t stream) {
  using namespace cfg;
  (void)in_sizes; (void)n_in; (void)out_size; (void)ws_size;

  const float* prev_d = (const float*)d_in[0];
  const float* x_obs  = (const float*)d_in[1];
  const float* a_prev = (const float*)d_in[2];
  const float* W1   = (const float*)d_in[3];
  const float* b1   = (const float*)d_in[4];
  const float* W2   = (const float*)d_in[5];
  const float* b2   = (const float*)d_in[6];
  const float* Wqkv = (const float*)d_in[7];
  const float* bqkv = (const float*)d_in[8];
  const float* Wo   = (const float*)d_in[9];
  const float* bo   = (const float*)d_in[10];
  const float* W_ih = (const float*)d_in[11];
  const float* b_ih = (const float*)d_in[12];
  const float* W_hh = (const float*)d_in[13];
  const float* b_hh = (const float*)d_in[14];

  char* ws = (char*)d_ws;
  size_t off = 0;
  auto carve = [&](size_t bytes) -> char* {
    char* p = ws + off;
    off += (bytes + 255) & ~(size_t)255;
    return p;
  };

  _Float16* Xh     = (_Float16*)carve((size_t)BS * IN * 2);   //  8 MB
  char*     R1     = carve((size_t)BS * D * 2);               // 32 MB: H1 -> ctx
  char*     R2     = carve((size_t)BS * D * 2);               // 32 MB: XA -> {reshh,gi,gh}
  char*     R3     = carve((size_t)BS * D3 * 2);              // 96 MB: QKV -> TO(f32)
  float*    ssum   = (float*)carve((size_t)H * BS * 4);       //  1 MB
  _Float16* prevdh = (_Float16*)carve((size_t)B * D * 2);
  _Float16* W1t    = (_Float16*)carve((size_t)IN * D * 2);
  _Float16* W2t    = (_Float16*)carve((size_t)D * (D - ACTN) * 2);
  _Float16* Wqkvt  = (_Float16*)carve((size_t)D * D3 * 2);
  _Float16* Wot    = (_Float16*)carve((size_t)D * D * 2);
  _Float16* Wiht   = (_Float16*)carve((size_t)KD * D3 * 2);
  _Float16* Whht   = (_Float16*)carve((size_t)D * D3 * 2);

  _Float16* H1    = (_Float16*)R1;
  _Float16* ctx   = (_Float16*)R1;                         // after H1 dead
  _Float16* XA    = (_Float16*)R2;
  _Float16* QKV   = (_Float16*)R3;
  float*    TO    = (float*)R3;                            // after QKV dead
  _Float16* reshh = (_Float16*)R2;                         // after XA dead
  float*    gi    = (float*)(R2 + (size_t)B * KD * 2);
  float*    gh    = (float*)(R2 + (size_t)B * KD * 2 + (size_t)B * D3 * 4);

  float* out_newd = (float*)d_out;
  float* out_resh = (float*)d_out + (size_t)B * D;

  // --- f32 -> f16 conversions / weight transposes ---
  cvt_f16_kernel<<<dim3((BS * IN + 255) / 256), 256, 0, stream>>>(x_obs, Xh, BS * IN);
  cvt_f16_kernel<<<dim3((B * D + 255) / 256), 256, 0, stream>>>(prev_d, prevdh, B * D);
  transpose_f16_kernel<<<dim3((IN * D + 255) / 256), 256, 0, stream>>>(W1, W1t, IN, D);
  transpose_f16_kernel<<<dim3((D * (D - ACTN) + 255) / 256), 256, 0, stream>>>(W2, W2t, D, D - ACTN);
  transpose_f16_kernel<<<dim3((D * D3 + 255) / 256), 256, 0, stream>>>(Wqkv, Wqkvt, D, D3);
  transpose_f16_kernel<<<dim3((D * D + 255) / 256), 256, 0, stream>>>(Wo, Wot, D, D);
  transpose_f16_kernel<<<dim3((KD * D3 + 255) / 256), 256, 0, stream>>>(W_ih, Wiht, KD, D3);
  transpose_f16_kernel<<<dim3((D * D3 + 255) / 256), 256, 0, stream>>>(W_hh, Whht, D, D3);

  // --- MLP: H1 = tanh(X @ W1 + b1); XA[:,0:248] = H1 @ W2 + b2; XA[:,248:] = a_prev ---
  gemm_f16_kernel<1, true, false><<<dim3(2, BS / 128), 256, 0, stream>>>(
      Xh, IN, W1t, IN, b1, H1, nullptr, D, BS, D, IN);
  gemm_f16_kernel<0, true, false><<<dim3(2, BS / 128), 256, 0, stream>>>(
      H1, D, W2t, D, b2, XA, nullptr, D, BS, D - ACTN, D);
  fill_aprev_kernel<<<dim3((BS * ACTN + 255) / 256), 256, 0, stream>>>(a_prev, XA);

  // --- QKV = XA @ Wqkv + bqkv ---
  gemm_f16_kernel<0, true, false><<<dim3(6, BS / 128), 256, 0, stream>>>(
      XA, D, Wqkvt, D, bqkv, QKV, nullptr, D3, BS, D3, D);

  // --- attention per (b,h) ---
  attn_kernel<<<dim3(B * H), 256, 0, stream>>>(QKV, ctx, ssum);

  // --- trans_out = ctx @ Wo + bo (fp32, reuses QKV region) ---
  gemm_f16_kernel<0, false, true><<<dim3(2, BS / 128), 256, 0, stream>>>(
      ctx, D, Wot, D, bo, nullptr, TO, D, BS, D, D);

  // --- top-k over summed scores + gather -> reshaped (output) + f16 copy ---
  topk_gather_kernel<<<dim3(B), 128, 0, stream>>>(ssum, TO, out_resh, reshh);

  // --- GRU GEMMs: gi = reshaped @ W_ih + b_ih ; gh = prev_d @ W_hh + b_hh ---
  gemm_f16_kernel<0, false, true><<<dim3(6, 4), 256, 0, stream>>>(
      reshh, KD, Wiht, KD, b_ih, nullptr, gi, D3, B, D3, KD);
  gemm_f16_kernel<0, false, true><<<dim3(6, 4), 256, 0, stream>>>(
      prevdh, D, Whht, D, b_hh, nullptr, gh, D3, B, D3, D);

  // --- GRU gates -> new_d (output) ---
  gru_kernel<<<dim3(B), 256, 0, stream>>>(gi, gh, prev_d, out_newd);
}
